// RKHS_VNN_19576460935813
// MI455X (gfx1250) — compile-verified, matrix operand order checked by reading, not run on hardware
//
#include <hip/hip_runtime.h>
#include <hip/hip_bf16.h>
#include <stdint.h>

// ---------------------------------------------------------------------------
// RKHS-VNN forward for MI455X (gfx1250, wave32, WMMA + async LDS staging).
//
// Roofline: eta (439 MB fp32, read once) dominates -> ~18.8 us at 23.3 TB/s.
// Per-voxel contraction [8 x 729] x [729 x 24] via v_wmma_f32_16x16x32_f16.
// eta is staged global -> LDS with GLOBAL_LOAD_ASYNC_TO_LDS_B128 (no VGPR
// round trip, ASYNCcnt-tracked); f32->f16 convert happens at B-fragment
// assembly. A-matrix padding rows are elided by clamping the row index
// (output rows b>=8 are never stored).
// ---------------------------------------------------------------------------

typedef __attribute__((ext_vector_type(16))) _Float16 v16h;
typedef __attribute__((ext_vector_type(8)))  _Float16 h8;
typedef __attribute__((ext_vector_type(8)))  float    v8f;
typedef __attribute__((ext_vector_type(4)))  float    v4f;
typedef __attribute__((ext_vector_type(4)))  int      v4i;

#define B_    8
#define C_    3
#define D_    8
#define H_    28
#define W_    28
#define P_    6272        // D*H*W
#define N_    27
#define NN_   729         // N*N
#define O_    24
#define CLS_  101
#define FEAT_ 150528      // O_*P_
#define TP_   16          // voxels per workgroup
#define KC_   32          // nm-chunk = WMMA K
#define NCH_  23          // ceil(729/32)
#define ESTR_ 33          // eta32 nm-stride pad (bank decorrelation)
#define FC_CHUNK 1024
#define NBLK3 (FEAT_ / FC_CHUNK)   // 147, exact

#if defined(__gfx1250__) && __has_builtin(__builtin_amdgcn_global_load_async_to_lds_b128)
#define HAVE_ASYNC 1
#else
#define HAVE_ASYNC 0
#endif

#define AS1 __attribute__((address_space(1)))
#define AS3 __attribute__((address_space(3)))

__device__ __forceinline__ void wait_async_all() {
#if __has_builtin(__builtin_amdgcn_s_wait_asynccnt)
    __builtin_amdgcn_s_wait_asynccnt(0);
#elif HAVE_ASYNC
    asm volatile("s_wait_asynccnt 0x0" ::: "memory");
#endif
}

// Stage one o-tile of the current nm-chunk: eta[o0+o][nm][p0..p0+15] -> LDS f32.
// 16-byte granules: lane moves 4 consecutive p. OOB nm clamps to a valid row
// (finite garbage x zero A-column == 0).
__device__ __forceinline__ void stage_eta(const float* __restrict__ eta,
                                          float* lds_eta, int p0, int ch,
                                          int o0, int numo, int tid) {
    const int total = numo * KC_ * 4;            // b128 granules
    for (int idx = tid; idx < total; idx += 256) {
        int pq  = idx & 3;
        int nml = (idx >> 2) & (KC_ - 1);
        int o   = idx >> 7;                      // KC_*4 == 128
        int nm  = ch * KC_ + nml;
        int gnm = nm < NN_ ? nm : NN_ - 1;
        long g  = (long)((o0 + o) * NN_ + gnm) * P_ + p0 + 4 * pq;
        float* dst = &lds_eta[(o * ESTR_ + nml) * TP_ + 4 * pq];
#if HAVE_ASYNC
        __builtin_amdgcn_global_load_async_to_lds_b128(
            (AS1 v4i*)(uintptr_t)(const void*)&eta[g],
            (AS3 v4i*)(uintptr_t)(void*)dst, 0, 0);
#else
        *(v4f*)dst = *(const v4f*)&eta[g];
#endif
        if (o0 == 0 && nm + KC_ < NN_)           // warm next chunk
            __builtin_prefetch(&eta[g + (long)KC_ * P_], 0, 1);
    }
}

// -------------------------------- Kernel 1 --------------------------------
// out_raw[b][o][p] = sum_{n,m} (1 + <x[b,:,n,p], x[b,:,m,p]>)^2 * eta[o,n,m,p]
__global__ __launch_bounds__(256)
void rkhs_wmma_kernel(const float* __restrict__ x,
                      const float* __restrict__ eta,
                      float* __restrict__ out_raw) {
    __shared__ alignas(16) _Float16 patch[B_ * C_ * N_ * TP_];   // 20.25 KB
    __shared__ alignas(16) _Float16 K_lds[TP_ * 8 * KC_];        // 8 KB
    __shared__ alignas(16) float    eta32[16 * ESTR_ * TP_];     // 33 KB

    const int tid  = threadIdx.x;
    const int p0   = blockIdx.x * TP_;
    const int lane = tid & 31;
    const int wid  = tid >> 5;
    const int r    = lane & 15;
    const int hi   = lane >> 4;
    const int h8off = hi ? 8 : 0;

    // One-time: 3x3x3 neighborhoods (zero-padded) for all (b,c) of 16 voxels.
    for (int idx = tid; idx < B_ * C_ * N_ * TP_; idx += 256) {
        int pl = idx & (TP_ - 1);
        int t2 = idx / TP_;
        int n  = t2 % N_;  t2 /= N_;
        int c  = t2 % C_;
        int b  = t2 / C_;
        int p  = p0 + pl;
        int d  = p / (H_ * W_);
        int rm = p - d * (H_ * W_);
        int h  = rm / W_;
        int w  = rm - h * W_;
        int dz = d + n / 9 - 1;
        int hy = h + (n / 3) % 3 - 1;
        int wx = w + n % 3 - 1;
        float v = 0.f;
        if ((unsigned)dz < D_ && (unsigned)hy < H_ && (unsigned)wx < W_)
            v = x[(((b * C_ + c) * D_ + dz) * H_ + hy) * W_ + wx];
        patch[idx] = (_Float16)v;
    }

    v8f acc[2][2];
#pragma unroll
    for (int v = 0; v < 2; ++v)
#pragma unroll
        for (int t = 0; t < 2; ++t)
#pragma unroll
            for (int e = 0; e < 8; ++e) acc[v][t][e] = 0.f;

    for (int ch = 0; ch < NCH_; ++ch) {
        __syncthreads();   // previous stage-1 fragment reads done; patches ready

        // Stage o-tile 0 (o = 0..15) asynchronously; overlap with K compute.
        stage_eta(eta, eta32, p0, ch, 0, 16, tid);

        // Degree-2 polynomial kernel chunk (f32 math, f16 store, 8 live rows).
        for (int idx = tid; idx < TP_ * 8 * KC_; idx += 256) {
            int nml = idx & (KC_ - 1);
            int t2  = idx >> 5;
            int b   = t2 & 7;
            int pl  = t2 >> 3;
            int nm  = ch * KC_ + nml;
            float kv = 0.f;
            if (nm < NN_) {
                int n = nm / N_;
                int m = nm - n * N_;
                float s = 1.f;
#pragma unroll
                for (int c = 0; c < C_; ++c)
                    s += (float)patch[((b * C_ + c) * N_ + n) * TP_ + pl]
                       * (float)patch[((b * C_ + c) * N_ + m) * TP_ + pl];
                kv = s * s;
            }
            K_lds[(pl * 8 + b) * KC_ + nml] = (_Float16)kv;
        }
        wait_async_all();
        __syncthreads();

        // A fragments (kept in regs across both o-tile stages).
        // Row M = r maps to batch b = r&7 (rows 8..15 duplicate, never stored).
        v16h afr[2];
#pragma unroll
        for (int v = 0; v < 2; ++v) {
            int pv    = 2 * wid + v;
            int abase = (pv * 8 + (r & 7)) * KC_ + h8off;
            h8 alo = *(const h8*)&K_lds[abase];
            h8 ahi = *(const h8*)&K_lds[abase + 16];
#pragma unroll
            for (int e = 0; e < 8; ++e) { afr[v][e] = alo[e]; afr[v][8 + e] = ahi[e]; }
        }

        // o-tile 0 WMMA.
#pragma unroll
        for (int v = 0; v < 2; ++v) {
            int pv = 2 * wid + v;
            int eb = (r * ESTR_ + h8off) * TP_ + pv;
            v16h bf;
#pragma unroll
            for (int e = 0; e < 8; ++e) {
                bf[e]     = (_Float16)eta32[eb + e * TP_];
                bf[8 + e] = (_Float16)eta32[eb + (16 + e) * TP_];
            }
            acc[v][0] = __builtin_amdgcn_wmma_f32_16x16x32_f16(
                false, afr[v], false, bf, (short)0, acc[v][0], false, false);
        }
        __syncthreads();   // tile-0 fragment reads done before restaging

        // Stage o-tile 1 (o = 16..23, 8 columns) into the same buffer.
        stage_eta(eta, eta32, p0, ch, 16, 8, tid);
        wait_async_all();
        __syncthreads();

#pragma unroll
        for (int v = 0; v < 2; ++v) {
            int pv = 2 * wid + v;
            int ol = r & 7;                      // cols 8..15 duplicate, unused
            int eb = (ol * ESTR_ + h8off) * TP_ + pv;
            v16h bf;
#pragma unroll
            for (int e = 0; e < 8; ++e) {
                bf[e]     = (_Float16)eta32[eb + e * TP_];
                bf[8 + e] = (_Float16)eta32[eb + (16 + e) * TP_];
            }
            acc[v][1] = __builtin_amdgcn_wmma_f32_16x16x32_f16(
                false, afr[v], false, bf, (short)0, acc[v][1], false, false);
        }
    }

    // C/D layout: lanes 0-15 hold M=0..7 (VGPR j -> M=j), N=lane. M>=8 is pad.
    if (lane < 16) {
#pragma unroll
        for (int v = 0; v < 2; ++v) {
            int p = p0 + 2 * wid + v;
#pragma unroll
            for (int t = 0; t < 2; ++t) {
                int o = 16 * t + lane;
                if (o < O_) {
#pragma unroll
                    for (int j = 0; j < 8; ++j)
                        out_raw[(j * O_ + o) * P_ + p] = acc[v][t][j];
                }
            }
        }
    }
}

// -------------------------------- Kernel 2 --------------------------------
// BatchNorm batch stats per channel, folded to scale/shift.
__global__ __launch_bounds__(256)
void bn_stats_kernel(const float* __restrict__ out_raw,
                     const float* __restrict__ gamma,
                     const float* __restrict__ beta,
                     float* __restrict__ scale,
                     float* __restrict__ shift) {
    __shared__ float s1[256];
    __shared__ float s2[256];
    const int o = blockIdx.x, tid = threadIdx.x;
    float sum = 0.f, ss = 0.f;
    for (int p = tid; p < P_; p += 256)
        for (int b = 0; b < B_; ++b) {
            float v = out_raw[(b * O_ + o) * P_ + p];
            sum += v; ss += v * v;
        }
    s1[tid] = sum; s2[tid] = ss;
    __syncthreads();
    for (int off = 128; off > 0; off >>= 1) {
        if (tid < off) { s1[tid] += s1[tid + off]; s2[tid] += s2[tid + off]; }
        __syncthreads();
    }
    if (tid == 0) {
        const float inv_n = 1.f / (float)(B_ * P_);
        float mean = s1[0] * inv_n;
        float var  = s2[0] * inv_n - mean * mean;
        float sc   = gamma[o] * rsqrtf(var + 1e-5f);
        scale[o] = sc;
        shift[o] = beta[o] - mean * sc;
    }
}

// -------------------------------- Kernel 3 --------------------------------
// fc partials: each block owns one 1024-wide op-slice (W_fc streamed once,
// feats normalized into LDS once). Wave w handles classes k = w, w+8, ...
__global__ __launch_bounds__(256)
void fc_partial_kernel(const float* __restrict__ out_raw,
                       const float* __restrict__ scale,
                       const float* __restrict__ shift,
                       const float* __restrict__ Wfc,
                       float* __restrict__ partial) {
    __shared__ float feats[B_][FC_CHUNK];   // 32 KB
    const int tid  = threadIdx.x;
    const int base = blockIdx.x * FC_CHUNK;
    for (int idx = tid; idx < B_ * FC_CHUNK; idx += 256) {
        int i  = idx & (FC_CHUNK - 1);
        int b  = idx >> 10;
        int op = base + i;
        int o  = op / P_;
        int p  = op - o * P_;
        feats[b][i] = out_raw[(b * O_ + o) * P_ + p] * scale[o] + shift[o];
    }
    __syncthreads();
    const int lane = tid & 31, wid = tid >> 5;
    for (int k = wid; k < CLS_; k += 8) {
        float acc[B_];
#pragma unroll
        for (int b = 0; b < B_; ++b) acc[b] = 0.f;
        const float* wrow = Wfc + (size_t)k * FEAT_ + base;
        for (int i = lane; i < FC_CHUNK; i += 32) {
            float wv = wrow[i];
#pragma unroll
            for (int b = 0; b < B_; ++b) acc[b] += wv * feats[b][i];
        }
#pragma unroll
        for (int b = 0; b < B_; ++b)
            for (int off = 16; off > 0; off >>= 1)
                acc[b] += __shfl_down(acc[b], off, 32);
        if (lane == 0)
            for (int b = 0; b < B_; ++b)
                partial[((size_t)blockIdx.x * B_ + b) * CLS_ + k] = acc[b];
    }
}

// -------------------------------- Kernel 4 --------------------------------
// Deterministic fixed-order partial reduction + bias -> logits [8, 101].
__global__ __launch_bounds__(256)
void fc_finish_kernel(const float* __restrict__ partial,
                      const float* __restrict__ bfc,
                      float* __restrict__ out) {
    int tid = blockIdx.x * blockDim.x + threadIdx.x;
    if (tid >= B_ * CLS_) return;
    int b = tid / CLS_;
    int k = tid - b * CLS_;
    float s = bfc[k];
    for (int blk = 0; blk < NBLK3; ++blk)
        s += partial[((size_t)blk * B_ + b) * CLS_ + k];
    out[b * CLS_ + k] = s;
}

// -------------------------------- Launch ----------------------------------
extern "C" void kernel_launch(void* const* d_in, const int* in_sizes, int n_in,
                              void* d_out, int out_size, void* d_ws, size_t ws_size,
                              hipStream_t stream) {
    const float* x     = (const float*)d_in[0];
    const float* eta   = (const float*)d_in[1];
    const float* gamma = (const float*)d_in[2];
    const float* beta  = (const float*)d_in[3];
    const float* Wfc   = (const float*)d_in[4];
    const float* bfc   = (const float*)d_in[5];
    float* out = (float*)d_out;

    // Workspace layout (floats): out_raw | scale | shift | fc partials (~5.3 MB)
    float* ws       = (float*)d_ws;
    float* out_raw  = ws;                                   // B*O*P = 1,204,224
    float* scale    = out_raw + (size_t)B_ * O_ * P_;       // 24
    float* shift    = scale + O_;                           // 24
    float* partial  = shift + O_;                           // 147*8*101 = 118,776

    rkhs_wmma_kernel<<<P_ / TP_, 256, 0, stream>>>(x, eta, out_raw);           // 392 WGs
    bn_stats_kernel<<<O_, 256, 0, stream>>>(out_raw, gamma, beta, scale, shift);
    fc_partial_kernel<<<NBLK3, 256, 0, stream>>>(out_raw, scale, shift, Wfc, partial);
    fc_finish_kernel<<<(B_ * CLS_ + 255) / 256, 256, 0, stream>>>(partial, bfc, out);
}
